// HeteroGAT_Transformer_Block_46196668235863
// MI455X (gfx1250) — compile-verified, matrix operand order checked by reading, not run on hardware
//
#include <hip/hip_runtime.h>
#include <stdint.h>

#define D      128
#define SEQL   30
#define NNODE  2000
#define NTOK   (NNODE * SEQL)
#define FFD    2048
#define NH     4
#define HD     32
#define EMM    64000
#define EMMSL  (EMM + NNODE)
#define EBM    2000

typedef __attribute__((ext_vector_type(16))) __bf16 v16bf;
typedef __attribute__((ext_vector_type(8)))  float  v8f;

union FragB { uint4 q[2]; v16bf v; };

__device__ __forceinline__ unsigned short f2bf(float f) {
  unsigned u = __float_as_uint(f);
  unsigned r = u + 0x7FFFu + ((u >> 16) & 1u);   // RNE
  return (unsigned short)(r >> 16);
}
__device__ __forceinline__ float bf2f(unsigned short h) {
  return __uint_as_float(((unsigned)h) << 16);
}
__device__ __forceinline__ unsigned fkey(float f) {  // order-preserving float->uint
  unsigned u = __float_as_uint(f);
  return (u & 0x80000000u) ? ~u : (u | 0x80000000u);
}
__device__ __forceinline__ float funkey(unsigned k) {
  unsigned u = (k & 0x80000000u) ? (k & 0x7FFFFFFFu) : ~k;
  return __uint_as_float(u);
}

// ---------------------------------------------------------------------------
// Generic bf16 WMMA GEMM: C[M,N] = A[M,K] @ W[N,K]^T + bias.  One wave per
// 16x16 tile, f32 accumulation, optional f32 and/or bf16 outputs.
// ---------------------------------------------------------------------------
__global__ __launch_bounds__(128) void gemm_bf16(
    const unsigned short* __restrict__ A, const unsigned short* __restrict__ W,
    const float* __restrict__ bias, float* __restrict__ Cf,
    unsigned short* __restrict__ Cb, int M, int N, int K, int relu)
{
  int lane = threadIdx.x;
  int mt = blockIdx.y * blockDim.y + threadIdx.y;
  if (mt * 16 >= M) return;
  int nt = blockIdx.x;
  int l15 = lane & 15;
  int khalf = (lane >> 4) << 3;     // lanes 16..31 hold K-offset +8 per ISA layout
  const unsigned short* ap = A + (size_t)(mt * 16 + l15) * K + khalf;
  const unsigned short* wp = W + (size_t)(nt * 16 + l15) * K + khalf;
  v8f acc = {0.f, 0.f, 0.f, 0.f, 0.f, 0.f, 0.f, 0.f};
  for (int k0 = 0; k0 < K; k0 += 32) {
    __builtin_prefetch(ap + 64, 0, 1);
    FragB fa, fb;
    fa.q[0] = *(const uint4*)(ap);
    fa.q[1] = *(const uint4*)(ap + 16);
    fb.q[0] = *(const uint4*)(wp);
    fb.q[1] = *(const uint4*)(wp + 16);
    ap += 32; wp += 32;
    acc = __builtin_amdgcn_wmma_f32_16x16x32_bf16(false, fa.v, false, fb.v,
                                                  (short)0, acc, false, false);
  }
  int ccol = nt * 16 + l15;
  int crow = mt * 16 + ((lane >> 4) << 3);
  float bv = bias ? bias[ccol] : 0.f;
#pragma unroll
  for (int i = 0; i < 8; ++i) {
    float v = acc[i] + bv;
    if (relu) v = fmaxf(v, 0.f);
    size_t idx = (size_t)(crow + i) * N + ccol;
    if (Cf) Cf[idx] = v;
    if (Cb) Cb[idx] = f2bf(v);
  }
}

// ---------------------------------------------------------------------------
// Fused FFN: O[16,128] = relu(X[16,128] @ W1^T + b1) @ W2^T + b2 per wave,
// never materializing the 60000x2048 intermediate (saves ~500MB HBM traffic).
// The 16x32 hidden tile goes C-layout -> LDS(bf16) -> A-fragment layout.
// ---------------------------------------------------------------------------
__global__ __launch_bounds__(128) void ffn_fused(
    const unsigned short* __restrict__ X, const unsigned short* __restrict__ W1,
    const float* __restrict__ B1, const unsigned short* __restrict__ W2,
    const float* __restrict__ B2, float* __restrict__ O, int M)
{
  __shared__ unsigned short hbuf[4 * 16 * 32];
  int lane = threadIdx.x;
  int strip = blockIdx.x * blockDim.y + threadIdx.y;
  if (strip * 16 >= M) return;
  unsigned short* lds = hbuf + threadIdx.y * (16 * 32);
  int l15 = lane & 15;
  int khalf = (lane >> 4) << 3;
  int rbase = (lane >> 4) << 3;

  FragB fa[4];                                 // whole 16x128 A strip in regs
  {
    const unsigned short* ap = X + (size_t)(strip * 16 + l15) * D + khalf;
#pragma unroll
    for (int kk = 0; kk < 4; ++kk) {
      fa[kk].q[0] = *(const uint4*)(ap + kk * 32);
      fa[kk].q[1] = *(const uint4*)(ap + kk * 32 + 16);
    }
  }
  v8f out[8];
#pragma unroll
  for (int i = 0; i < 8; ++i) out[i] = v8f{0.f,0.f,0.f,0.f,0.f,0.f,0.f,0.f};

  for (int hc = 0; hc < FFD / 32; ++hc) {
#pragma unroll
    for (int t = 0; t < 2; ++t) {              // two 16-wide hidden tiles
      int hrow = hc * 32 + t * 16 + l15;
      const unsigned short* w1p = W1 + (size_t)hrow * D + khalf;
      v8f h = v8f{0.f,0.f,0.f,0.f,0.f,0.f,0.f,0.f};
#pragma unroll
      for (int kk = 0; kk < 4; ++kk) {
        FragB fw;
        fw.q[0] = *(const uint4*)(w1p + kk * 32);
        fw.q[1] = *(const uint4*)(w1p + kk * 32 + 16);
        h = __builtin_amdgcn_wmma_f32_16x16x32_bf16(false, fa[kk].v, false, fw.v,
                                                    (short)0, h, false, false);
      }
      float b1 = B1[hrow];
#pragma unroll
      for (int i = 0; i < 8; ++i)              // relu + C-layout -> LDS
        lds[(rbase + i) * 32 + t * 16 + l15] = f2bf(fmaxf(h[i] + b1, 0.f));
    }
    FragB fh;                                   // reload as A fragment
    fh.q[0] = *(const uint4*)(lds + l15 * 32 + khalf);
    fh.q[1] = *(const uint4*)(lds + l15 * 32 + khalf + 16);
#pragma unroll
    for (int nt = 0; nt < 8; ++nt) {
      const unsigned short* w2p = W2 + (size_t)(nt * 16 + l15) * FFD + hc * 32 + khalf;
      FragB fw2;
      fw2.q[0] = *(const uint4*)(w2p);
      fw2.q[1] = *(const uint4*)(w2p + 16);
      out[nt] = __builtin_amdgcn_wmma_f32_16x16x32_bf16(false, fh.v, false, fw2.v,
                                                        (short)0, out[nt], false, false);
    }
  }
#pragma unroll
  for (int nt = 0; nt < 8; ++nt) {
    int ccol = nt * 16 + l15;
    float b2 = B2[ccol];
#pragma unroll
    for (int i = 0; i < 8; ++i)
      O[(size_t)(strip * 16 + rbase + i) * D + ccol] = out[nt][i] + b2;
  }
}

// ---------------------------------------------------------------------------
// Attention: one wave per (head, node); 30x30 scores in registers, k/v in LDS.
// ---------------------------------------------------------------------------
__global__ __launch_bounds__(32) void attn_kernel(
    const unsigned short* __restrict__ qkv, unsigned short* __restrict__ out)
{
  __shared__ float kS[SEQL * HD];
  __shared__ float vS[SEQL * HD];
  int h = blockIdx.x, n = blockIdx.y, lane = threadIdx.x;
  size_t tb = (size_t)n * SEQL;
  for (int idx = lane; idx < SEQL * HD; idx += 32) {
    int r = idx >> 5, c = idx & 31;
    kS[idx] = bf2f(qkv[(tb + r) * 384 + 128 + h * HD + c]);
    vS[idx] = bf2f(qkv[(tb + r) * 384 + 256 + h * HD + c]);
  }
  __syncthreads();
  if (lane >= SEQL) return;
  float q[HD];
#pragma unroll
  for (int c = 0; c < HD; ++c) q[c] = bf2f(qkv[(tb + lane) * 384 + h * HD + c]);
  float sc[SEQL];
  const float scale = 0.17677669529663687f;    // 1/sqrt(32)
  float mx = -1e30f;
  for (int j = 0; j < SEQL; ++j) {
    float s = 0.f;
#pragma unroll
    for (int c = 0; c < HD; ++c) s += q[c] * kS[j * HD + c];
    s *= scale; sc[j] = s; mx = fmaxf(mx, s);
  }
  float psum = 0.f;
  for (int j = 0; j < SEQL; ++j) { float p = __expf(sc[j] - mx); sc[j] = p; psum += p; }
  float inv = 1.f / psum;
  float o[HD];
#pragma unroll
  for (int c = 0; c < HD; ++c) o[c] = 0.f;
  for (int j = 0; j < SEQL; ++j) {
    float p = sc[j];
#pragma unroll
    for (int c = 0; c < HD; ++c) o[c] += p * vS[j * HD + c];
  }
  for (int c = 0; c < HD; ++c)
    out[(tb + lane) * D + h * HD + c] = f2bf(o[c] * inv);
}

// ---------------------------------------------------------------------------
// Residual + LayerNorm over D=128, wave32 shuffle + 4-wave shared reduction.
// ---------------------------------------------------------------------------
__global__ __launch_bounds__(128) void add_ln(
    const float* __restrict__ X, const float* __restrict__ R,
    const float* __restrict__ G, const float* __restrict__ Bt,
    float* __restrict__ OF, unsigned short* __restrict__ OB)
{
  __shared__ float s1[4], s2[4];
  int row = blockIdx.x, t = threadIdx.x;
  float v = X[(size_t)row * D + t] + R[(size_t)row * D + t];
  float a = v, b = v * v;
#pragma unroll
  for (int off = 16; off > 0; off >>= 1) {
    a += __shfl_xor(a, off, 32);
    b += __shfl_xor(b, off, 32);
  }
  if ((t & 31) == 0) { s1[t >> 5] = a; s2[t >> 5] = b; }
  __syncthreads();
  float sum = s1[0] + s1[1] + s1[2] + s1[3];
  float sq  = s2[0] + s2[1] + s2[2] + s2[3];
  float mean = sum * (1.f / D);
  float var  = sq * (1.f / D) - mean * mean;
  float y = (v - mean) * rsqrtf(var + 1e-5f) * G[t] + Bt[t];
  OF[(size_t)row * D + t] = y;
  if (OB) OB[(size_t)row * D + t] = f2bf(y);
}

__global__ __launch_bounds__(128) void fuse_mean(
    const float* __restrict__ ctx, const float* __restrict__ gnn,
    const float* __restrict__ G, const float* __restrict__ Bt,
    float* __restrict__ out)
{
  __shared__ float s1[4], s2[4];
  int n = blockIdx.x, t = threadIdx.x;
  float v = ctx[(size_t)n * D + t] + gnn[(size_t)n * D + t];
  float a = v, b = v * v;
#pragma unroll
  for (int off = 16; off > 0; off >>= 1) {
    a += __shfl_xor(a, off, 32);
    b += __shfl_xor(b, off, 32);
  }
  if ((t & 31) == 0) { s1[t >> 5] = a; s2[t >> 5] = b; }
  __syncthreads();
  float sum = s1[0] + s1[1] + s1[2] + s1[3];
  float sq  = s2[0] + s2[1] + s2[2] + s2[3];
  float mean = sum * (1.f / D);
  float var  = sq * (1.f / D) - mean * mean;
  float y = (v - mean) * rsqrtf(var + 1e-5f) * G[t] + Bt[t];
  atomicAdd(&out[t], y * (1.f / NNODE));
}

// ---------------------------------------------------------------------------
// Embedding: x = hist @ emb_w^T + emb_b + pos (K=16, VALU is fine here).
// ---------------------------------------------------------------------------
__global__ __launch_bounds__(128) void embed_kernel(
    const float* __restrict__ hist, const float* __restrict__ Wt,
    const float* __restrict__ b, const float* __restrict__ pos,
    float* __restrict__ XF, unsigned short* __restrict__ XB)
{
  int t = blockIdx.x, d = threadIdx.x;
  const float* hp = hist + (size_t)t * 16;
  float s = b[d] + pos[(t % SEQL) * D + d];
#pragma unroll
  for (int f = 0; f < 16; ++f) s += hp[f] * Wt[d * 16 + f];
  XF[(size_t)t * D + d] = s;
  XB[(size_t)t * D + d] = f2bf(s);
}

__global__ __launch_bounds__(128) void ctx_gather(
    const float* __restrict__ X, float* __restrict__ CF, unsigned short* __restrict__ CB)
{
  int n = blockIdx.x, d = threadIdx.x;
  float v = X[(size_t)(n * SEQL + SEQL - 1) * D + d];
  CF[(size_t)n * D + d] = v;
  CB[(size_t)n * D + d] = f2bf(v);
}

__global__ void conv_bf16_k(const float* __restrict__ in, unsigned short* __restrict__ o, int n) {
  int i = blockIdx.x * blockDim.x + threadIdx.x;
  if (i < n) o[i] = f2bf(in[i]);
}
__global__ void zero_k(unsigned* p, int n) {
  int i = blockIdx.x * blockDim.x + threadIdx.x;
  if (i < n) p[i] = 0u;
}

// ---------------------------------------------------------------------------
// GATv2 edge kernels (wave per edge). xlvec != nullptr => source features are
// the constant lin_l bias vector (buffer->machine relation with zero inputs).
// ---------------------------------------------------------------------------
__global__ __launch_bounds__(128) void gat_logits(
    const float* __restrict__ xl, const float* __restrict__ xlvec,
    const float* __restrict__ xr, const float* __restrict__ att,
    const int* __restrict__ srci, const int* __restrict__ dsti,
    int E, int Eorig, unsigned* __restrict__ maxb, float* __restrict__ lg)
{
  int e = blockIdx.x * blockDim.y + threadIdx.y;
  if (e >= E) return;
  int lane = threadIdx.x;
  int s, d;
  if (e < Eorig) { s = srci[e]; d = dsti[e]; } else { s = e - Eorig; d = s; }
  float p0 = 0.f, p1 = 0.f;
#pragma unroll
  for (int cc = 0; cc < 4; ++cc) {
    int c = lane + cc * 32;
    float xl0 = xlvec ? xlvec[c]       : xl[(size_t)s * 256 + c];
    float xl1 = xlvec ? xlvec[128 + c] : xl[(size_t)s * 256 + 128 + c];
    float e0 = xl0 + xr[(size_t)d * 256 + c];
    float e1 = xl1 + xr[(size_t)d * 256 + 128 + c];
    e0 = e0 > 0.f ? e0 : 0.2f * e0;            // leaky_relu(0.2)
    e1 = e1 > 0.f ? e1 : 0.2f * e1;
    p0 += e0 * att[c];
    p1 += e1 * att[128 + c];
  }
#pragma unroll
  for (int off = 16; off > 0; off >>= 1) {
    p0 += __shfl_xor(p0, off, 32);
    p1 += __shfl_xor(p1, off, 32);
  }
  if (lane == 0) {
    lg[(size_t)e * 2]     = p0;
    lg[(size_t)e * 2 + 1] = p1;
    atomicMax(&maxb[d * 2],     fkey(p0));
    atomicMax(&maxb[d * 2 + 1], fkey(p1));
  }
}

__global__ void gat_mdecode(const unsigned* __restrict__ maxb, float* __restrict__ mv, int n) {
  int i = blockIdx.x * blockDim.x + threadIdx.x;
  if (i < n) { unsigned k = maxb[i]; mv[i] = (k == 0u) ? 0.f : funkey(k); }  // no-edge -> 0
}

__global__ void gat_expsum(float* __restrict__ lg, const float* __restrict__ mv,
                           const int* __restrict__ dsti, int E, int Eorig,
                           float* __restrict__ ssum)
{
  int i = blockIdx.x * blockDim.x + threadIdx.x;
  if (i >= E * 2) return;
  int e = i >> 1, h = i & 1;
  int d = (e < Eorig) ? dsti[e] : (e - Eorig);
  float ex = __expf(lg[i] - mv[d * 2 + h]);
  lg[i] = ex;
  atomicAdd(&ssum[d * 2 + h], ex);
}

__global__ __launch_bounds__(128) void gat_scatter(
    const float* __restrict__ lg, const float* __restrict__ ssum,
    const float* __restrict__ xl, const float* __restrict__ xlvec,
    const int* __restrict__ srci, const int* __restrict__ dsti,
    int E, int Eorig, float* __restrict__ out)
{
  int e = blockIdx.x * blockDim.y + threadIdx.y;
  if (e >= E) return;
  int lane = threadIdx.x;
  int s, d;
  if (e < Eorig) { s = srci[e]; d = dsti[e]; } else { s = e - Eorig; d = s; }
  float a0 = lg[(size_t)e * 2]     / (ssum[d * 2]     + 1e-16f);
  float a1 = lg[(size_t)e * 2 + 1] / (ssum[d * 2 + 1] + 1e-16f);
#pragma unroll
  for (int cc = 0; cc < 4; ++cc) {
    int c = lane + cc * 32;
    float xl0 = xlvec ? xlvec[c]       : xl[(size_t)s * 256 + c];
    float xl1 = xlvec ? xlvec[128 + c] : xl[(size_t)s * 256 + 128 + c];
    atomicAdd(&out[(size_t)d * 256 + c],       a0 * xl0);
    atomicAdd(&out[(size_t)d * 256 + 128 + c], a1 * xl1);
  }
}

__global__ void gnn_prep(const float* __restrict__ acc, const float* __restrict__ b1,
                         const float* __restrict__ b2, unsigned short* __restrict__ ob)
{
  int n = blockIdx.x, j = threadIdx.x;  // block 256
  ob[(size_t)n * 256 + j] = f2bf(acc[(size_t)n * 256 + j] + b1[j] + b2[j]);
}

// ---------------------------------------------------------------------------
extern "C" void kernel_launch(void* const* d_in, const int* in_sizes, int n_in,
                              void* d_out, int out_size, void* d_ws, size_t ws_size,
                              hipStream_t stream)
{
  (void)in_sizes; (void)n_in; (void)out_size; (void)ws_size;
  const float* hist  = (const float*)d_in[0];
  const float* emb_w = (const float*)d_in[1];
  const float* emb_b = (const float*)d_in[2];
  const float* pos   = (const float*)d_in[3];
  const float* qkv_w = (const float*)d_in[4];
  const float* qkv_b = (const float*)d_in[5];
  const float* out_w = (const float*)d_in[6];
  const float* out_b = (const float*)d_in[7];
  const float* ff1_w = (const float*)d_in[8];
  const float* ff1_b = (const float*)d_in[9];
  const float* ff2_w = (const float*)d_in[10];
  const float* ff2_b = (const float*)d_in[11];
  const float* ln1_g = (const float*)d_in[12];
  const float* ln1_b = (const float*)d_in[13];
  const float* ln2_g = (const float*)d_in[14];
  const float* ln2_b = (const float*)d_in[15];
  const float* gll_w = (const float*)d_in[16];
  const float* gll_b = (const float*)d_in[17];
  const float* glr_w = (const float*)d_in[18];
  const float* glr_b = (const float*)d_in[19];
  const float* gatt  = (const float*)d_in[20];
  const float* gbias = (const float*)d_in[21];
  const float* gnl_w = (const float*)d_in[22];
  const float* gnl_b = (const float*)d_in[23];
  const float* fn_g  = (const float*)d_in[24];
  const float* fn_b  = (const float*)d_in[25];
  const int*   e_mm  = (const int*)d_in[26];
  const int*   ebm_s = (const int*)d_in[29];
  const int*   ebm_d = (const int*)d_in[30];
  float* outp = (float*)d_out;

  char* ws = (char*)d_ws;
  size_t off = 0;
  auto alloc = [&](size_t bytes) -> void* {
    void* p = ws + off;
    off = (off + bytes + 255) & ~(size_t)255;
    return p;
  };
  auto cvt = [&](const float* src, unsigned short* dst, int n) {
    conv_bf16_k<<<(n + 255) / 256, 256, 0, stream>>>(src, dst, n);
  };
  auto gemm = [&](const unsigned short* A, const unsigned short* W, const float* bias,
                  float* Cf, unsigned short* Cb, int M, int N, int K) {
    dim3 grid(N / 16, (M / 16 + 3) / 4);
    gemm_bf16<<<grid, dim3(32, 4), 0, stream>>>(A, W, bias, Cf, Cb, M, N, K, 0);
  };

  // ---- bf16 weight copies ----
  unsigned short* qkvw_b = (unsigned short*)alloc(2 * 384 * D * 2);
  unsigned short* outw_b = (unsigned short*)alloc(2 * D * D * 2);
  unsigned short* ff1w_b = (unsigned short*)alloc(2 * FFD * D * 2);
  unsigned short* ff2w_b = (unsigned short*)alloc(2 * D * FFD * 2);
  unsigned short* lr1w_b = (unsigned short*)alloc(256 * D * 2);
  unsigned short* lr2w_b = (unsigned short*)alloc(256 * D * 2);
  unsigned short* ll2w_b = (unsigned short*)alloc(256 * D * 2);
  unsigned short* gnlw_b = (unsigned short*)alloc(D * 256 * 2);
  cvt(qkv_w, qkvw_b, 2 * 384 * D);
  cvt(out_w, outw_b, 2 * D * D);
  cvt(ff1_w, ff1w_b, 2 * FFD * D);
  cvt(ff2_w, ff2w_b, 2 * D * FFD);
  cvt(glr_w + 1 * 256 * D, lr1w_b, 256 * D);
  cvt(glr_w + 2 * 256 * D, lr2w_b, 256 * D);
  cvt(gll_w + 2 * 256 * D, ll2w_b, 256 * D);
  cvt(gnl_w, gnlw_b, D * 256);

  // ---- activations ----
  float*          xf    = (float*)alloc((size_t)NTOK * D * 4);
  unsigned short* xb    = (unsigned short*)alloc((size_t)NTOK * D * 2);
  unsigned short* qkvb  = (unsigned short*)alloc((size_t)NTOK * 384 * 2);
  unsigned short* attnb = (unsigned short*)alloc((size_t)NTOK * D * 2);
  float*          tmpf  = (float*)alloc((size_t)NTOK * D * 4);
  float*          ctxf  = (float*)alloc((size_t)NNODE * D * 4);
  unsigned short* ctxb  = (unsigned short*)alloc((size_t)NNODE * D * 2);
  float*          xr_mm = (float*)alloc((size_t)NNODE * 256 * 4);
  float*          xl_mm = (float*)alloc((size_t)NNODE * 256 * 4);
  float*          xr_bm = (float*)alloc((size_t)NNODE * 256 * 4);
  unsigned*       mb_mm = (unsigned*)alloc(NNODE * 2 * 4);
  unsigned*       mb_bm = (unsigned*)alloc(NNODE * 2 * 4);
  float*          mv_mm = (float*)alloc(NNODE * 2 * 4);
  float*          mv_bm = (float*)alloc(NNODE * 2 * 4);
  float*          ss_mm = (float*)alloc(NNODE * 2 * 4);
  float*          ss_bm = (float*)alloc(NNODE * 2 * 4);
  float*          lg_mm = (float*)alloc((size_t)EMMSL * 2 * 4);
  float*          lg_bm = (float*)alloc((size_t)EBM * 2 * 4);
  float*          oacc  = (float*)alloc((size_t)NNODE * 256 * 4);
  unsigned short* gnninb= (unsigned short*)alloc((size_t)NNODE * 256 * 2);
  float*          gnnf  = (float*)alloc((size_t)NNODE * D * 4);

  // ---- embedding ----
  embed_kernel<<<NTOK, 128, 0, stream>>>(hist, emb_w, emb_b, pos, xf, xb);

  // ---- transformer layers ----
  for (int l = 0; l < 2; ++l) {
    gemm(xb, qkvw_b + (size_t)l * 384 * D, qkv_b + l * 384, nullptr, qkvb, NTOK, 384, D);
    attn_kernel<<<dim3(NH, NNODE), 32, 0, stream>>>(qkvb, attnb);
    gemm(attnb, outw_b + (size_t)l * D * D, out_b + l * D, tmpf, nullptr, NTOK, D, D);
    add_ln<<<NTOK, 128, 0, stream>>>(xf, tmpf, ln1_g + l * D, ln1_b + l * D, xf, xb);
    ffn_fused<<<(NTOK / 16 + 3) / 4, dim3(32, 4), 0, stream>>>(
        xb, ff1w_b + (size_t)l * FFD * D, ff1_b + l * FFD,
        ff2w_b + (size_t)l * D * FFD, ff2_b + l * D, tmpf, NTOK);
    add_ln<<<NTOK, 128, 0, stream>>>(xf, tmpf, ln2_g + l * D, ln2_b + l * D, xf, xb);
  }

  // ---- context + GAT linear projections ----
  ctx_gather<<<NNODE, 128, 0, stream>>>(xf, ctxf, ctxb);
  gemm(ctxb, lr2w_b, glr_b + 2 * 256, xr_mm, nullptr, NNODE, 256, D);
  gemm(ctxb, ll2w_b, gll_b + 2 * 256, xl_mm, nullptr, NNODE, 256, D);
  gemm(ctxb, lr1w_b, glr_b + 1 * 256, xr_bm, nullptr, NNODE, 256, D);

  // ---- zero atomic accumulators + output ----
  zero_k<<<(NNODE * 2 + 255) / 256, 256, 0, stream>>>(mb_mm, NNODE * 2);
  zero_k<<<(NNODE * 2 + 255) / 256, 256, 0, stream>>>(mb_bm, NNODE * 2);
  zero_k<<<(NNODE * 2 + 255) / 256, 256, 0, stream>>>((unsigned*)ss_mm, NNODE * 2);
  zero_k<<<(NNODE * 2 + 255) / 256, 256, 0, stream>>>((unsigned*)ss_bm, NNODE * 2);
  zero_k<<<(NNODE * 256 + 255) / 256, 256, 0, stream>>>((unsigned*)oacc, NNODE * 256);
  zero_k<<<1, 128, 0, stream>>>((unsigned*)outp, D);

  const float* xlvec_bm = gll_b + 1 * 256;  // lin_l(zeros) == bias, for every edge

  // ---- buffer -> machine relation ----
  gat_logits<<<(EBM + 3) / 4, dim3(32, 4), 0, stream>>>(
      nullptr, xlvec_bm, xr_bm, gatt + 1 * 256, ebm_s, ebm_d, EBM, EBM, mb_bm, lg_bm);
  gat_mdecode<<<(NNODE * 2 + 255) / 256, 256, 0, stream>>>(mb_bm, mv_bm, NNODE * 2);
  gat_expsum<<<(EBM * 2 + 255) / 256, 256, 0, stream>>>(lg_bm, mv_bm, ebm_d, EBM, EBM, ss_bm);
  gat_scatter<<<(EBM + 3) / 4, dim3(32, 4), 0, stream>>>(
      lg_bm, ss_bm, nullptr, xlvec_bm, ebm_s, ebm_d, EBM, EBM, oacc);

  // ---- machine -> machine relation (with appended self loops) ----
  gat_logits<<<(EMMSL + 3) / 4, dim3(32, 4), 0, stream>>>(
      xl_mm, nullptr, xr_mm, gatt + 2 * 256, e_mm, e_mm + EMM, EMMSL, EMM, mb_mm, lg_mm);
  gat_mdecode<<<(NNODE * 2 + 255) / 256, 256, 0, stream>>>(mb_mm, mv_mm, NNODE * 2);
  gat_expsum<<<(EMMSL * 2 + 255) / 256, 256, 0, stream>>>(lg_mm, mv_mm, e_mm + EMM, EMMSL, EMM, ss_mm);
  gat_scatter<<<(EMMSL + 3) / 4, dim3(32, 4), 0, stream>>>(
      lg_mm, ss_mm, xl_mm, nullptr, e_mm, e_mm + EMM, EMMSL, EMM, oacc);

  // ---- gnn linear + fuse + mean ----
  gnn_prep<<<NNODE, 256, 0, stream>>>(oacc, gbias + 1 * 256, gbias + 2 * 256, gnninb);
  gemm(gnninb, gnlw_b, gnl_b, gnnf, nullptr, NNODE, D, 256);
  fuse_mean<<<NNODE, 128, 0, stream>>>(ctxf, gnnf, fn_g, fn_b, outp);
}
